// WaveRNN_54142357734073
// MI455X (gfx1250) — matrix operand order repaired and checked by reading, not compiled, not run on hardware
//
#include <hip/hip_runtime.h>

// ---------------------------------------------------------------------------
// WaveRNN GRU step for MI455X (gfx1250): bf16 WMMA GEMMs + f32 epilogues.
// B=16384, H=896 (S=448), Q=256.  All GEMM K dims are multiples of 32.
// v2: 32-row register blocking (halves weight-fragment traffic) and
//     software-pipelined fragment loads (overlaps vmem with v_wmma).
// ---------------------------------------------------------------------------

typedef __bf16 bf16;
typedef __attribute__((ext_vector_type(16))) __bf16 bf16x16;
typedef __attribute__((ext_vector_type(8)))  float  floatx8;

constexpr int H  = 896;
constexpr int S  = 448;     // H/2
constexpr int Q  = 256;
constexpr int B  = 16384;
constexpr int H3 = 3 * H;

// ---- WMMA helpers ---------------------------------------------------------

__device__ __forceinline__ floatx8 wmma_bf16(bf16x16 a, bf16x16 b, floatx8 c) {
  // v_wmma_f32_16x16x32_bf16: D = A(16x32) * B(32x16) + C(16x16 f32)
  return __builtin_amdgcn_wmma_f32_16x16x32_bf16(
      /*neg_a=*/false, a, /*neg_b=*/false, b,
      /*c_mod=*/(short)0, c, /*reuse_a=*/false, /*reuse_b=*/false);
}

// A-matrix 16x32 bf16 fragment. p points at (row m, K=kk) in a row-major
// bf16 matrix. Lane half (lane>>4) selects the K sub-stripe:
//   V0-3: K = kb+0..7 ; V4-7: K = 16+kb+0..7 ; kb = half*8
__device__ __forceinline__ bf16x16 load_a_frag(const bf16* __restrict__ p, int half) {
  const int kb = half << 3;
  union { bf16x16 v; uint4 q[2]; } u;
  u.q[0] = *reinterpret_cast<const uint4*>(p + kb);
  u.q[1] = *reinterpret_cast<const uint4*>(p + 16 + kb);
  return u.v;
}

// B-matrix 32x16 bf16 fragment, same K order as A: elements 0-7 hold
// K = kb+0..7, elements 8-15 hold K = 16+kb+0..7, kb = half*8.  p points at (row n of the weight matrix, K=kk) — weights are
// stored [N,K] row-major, i.e. B^T, which is exactly what "@ W.T" needs.
__device__ __forceinline__ bf16x16 load_b_frag(const bf16* __restrict__ p, int half) {
  const int ks = half << 3;
  union { bf16x16 v; uint4 q[2]; } u;
  u.q[0] = *reinterpret_cast<const uint4*>(p + ks);
  u.q[1] = *reinterpret_cast<const uint4*>(p + 16 + ks);
  return u.v;
}

__device__ __forceinline__ float sigmoid_f(float x) {
  return 1.0f / (1.0f + __expf(-x));
}

// ---- Kernel 0: f32 -> bf16 conversion (vectorized x4) ---------------------

__global__ void cvt_f32_to_bf16(const float* __restrict__ in,
                                bf16* __restrict__ out, int n4) {
  int i = blockIdx.x * blockDim.x + threadIdx.x;
  if (i >= n4) return;
  float4 v = reinterpret_cast<const float4*>(in)[i];
  bf16 o[4] = { (bf16)v.x, (bf16)v.y, (bf16)v.z, (bf16)v.w };
  reinterpret_cast<uint2*>(out)[i] = *reinterpret_cast<uint2*>(o);
}

// ---- GRU elementwise epilogue for one 16x16 accumulator trio --------------

__device__ __forceinline__ void gru_epilogue(
    const floatx8& aU, const floatx8& aR, const floatx8& aE,
    int mbase, int h,
    float wu0, float wu1, float wu2,
    float wr0, float wr1, float wr2,
    float we0, float we1, float we2,
    float vbu, float vbr, float vbe,
    const float* __restrict__ prev_y, const float* __restrict__ coarse,
    const float* __restrict__ prev_hidden,
    float* __restrict__ hidden_f32, bf16* __restrict__ hidden_bf)
{
#pragma unroll
  for (int i = 0; i < 8; ++i) {
    const int b = mbase + i;
    const float py0 = prev_y[2 * b], py1 = prev_y[2 * b + 1], cc = coarse[b];
    const float Iu = wu0 * py0 + wu1 * py1 + wu2 * cc;
    const float Ir = wr0 * py0 + wr1 * py1 + wr2 * cc;
    const float Ie = we0 * py0 + we1 * py1 + we2 * cc;
    const float u = sigmoid_f(aU[i] + Iu + vbu);
    const float r = sigmoid_f(aR[i] + Ir + vbr);
    const float e = tanhf(r * aE[i] + Ie + vbe);
    const float ph  = prev_hidden[(size_t)b * H + h];
    const float hid = u * ph + (1.0f - u) * e;
    hidden_f32[(size_t)b * H + h] = hid;
    hidden_bf[(size_t)b * H + h]  = (bf16)hid;
  }
}

// ---- Kernel 1: fused GRU update -------------------------------------------
// One wave per 32(batch) x 16(h) tile: 2 A frags x 3 gate B frags = 6 WMMAs
// per K step, B fragments reused across both row halves.  Fragment loads for
// step kk+32 are issued before the kk WMMAs (software pipeline).
// grid = (H/16, B/32), block = 32.

__global__ void __launch_bounds__(32)
gru_wmma_kernel(const bf16* __restrict__ Abf,     // prev_hidden bf16 [B,H]
                const bf16* __restrict__ Rbf,     // R_w bf16 [3H,H]
                const float* __restrict__ prev_y, // [B,2]
                const float* __restrict__ coarse, // [B,1]
                const float* __restrict__ prev_hidden, // f32 [B,H]
                const float* __restrict__ Ic_w,   // [3S,2]
                const float* __restrict__ If_w,   // [3S,3]
                const float* __restrict__ bu,
                const float* __restrict__ br,
                const float* __restrict__ be,
                float* __restrict__ hidden_f32,   // -> d_out region [B,H]
                bf16* __restrict__ hidden_bf)     // ws [B,H]
{
  const int lane = threadIdx.x;
  const int half = lane >> 4;
  const int lid  = lane & 15;
  const int th   = blockIdx.x;           // h tile: 0..55
  const int tb   = blockIdx.y;           // 32-row batch tile: 0..511
  const int h    = th * 16 + lid;        // output column / weight row

  const bf16* arow0 = Abf + (size_t)(tb * 32 + lid) * H;
  const bf16* arow1 = arow0 + (size_t)16 * H;
  const bf16* bwu   = Rbf + (size_t)h * H;
  const bf16* bwr   = Rbf + (size_t)(H + h) * H;
  const bf16* bwe   = Rbf + (size_t)(2 * H + h) * H;

  floatx8 accU0 = {}, accU1 = {}, accR0 = {}, accR1 = {}, accE0 = {}, accE1 = {};

  // Pipeline: preload kk=0 fragments, then load kk+32 before issuing kk WMMAs.
  bf16x16 a0 = load_a_frag(arow0, half);
  bf16x16 a1 = load_a_frag(arow1, half);
  bf16x16 fu = load_b_frag(bwu, half);
  bf16x16 fr = load_b_frag(bwr, half);
  bf16x16 fe = load_b_frag(bwe, half);

  for (int kk = 32; kk < H; kk += 32) {
    bf16x16 na0 = load_a_frag(arow0 + kk, half);
    bf16x16 na1 = load_a_frag(arow1 + kk, half);
    bf16x16 nfu = load_b_frag(bwu + kk, half);
    bf16x16 nfr = load_b_frag(bwr + kk, half);
    bf16x16 nfe = load_b_frag(bwe + kk, half);

    accU0 = wmma_bf16(a0, fu, accU0);
    accU1 = wmma_bf16(a1, fu, accU1);
    accR0 = wmma_bf16(a0, fr, accR0);
    accR1 = wmma_bf16(a1, fr, accR1);
    accE0 = wmma_bf16(a0, fe, accE0);
    accE1 = wmma_bf16(a1, fe, accE1);

    a0 = na0; a1 = na1; fu = nfu; fr = nfr; fe = nfe;
  }
  accU0 = wmma_bf16(a0, fu, accU0);
  accU1 = wmma_bf16(a1, fu, accU1);
  accR0 = wmma_bf16(a0, fr, accR0);
  accR1 = wmma_bf16(a1, fr, accR1);
  accE0 = wmma_bf16(a0, fe, accE0);
  accE1 = wmma_bf16(a1, fe, accE1);

  // Input-projection weights for this column h (wave-uniform branch: S%16==0).
  float wu0, wu1, wu2, wr0, wr1, wr2, we0, we1, we2;
  if (h < S) {                           // coarse half: prev_y @ Ic_w.T
    wu0 = Ic_w[2 * h];           wu1 = Ic_w[2 * h + 1];           wu2 = 0.0f;
    wr0 = Ic_w[2 * (S + h)];     wr1 = Ic_w[2 * (S + h) + 1];     wr2 = 0.0f;
    we0 = Ic_w[2 * (2 * S + h)]; we1 = Ic_w[2 * (2 * S + h) + 1]; we2 = 0.0f;
  } else {                               // fine half: [prev_y,cc] @ If_w.T
    const int hf = h - S;
    wu0 = If_w[3 * hf];      wu1 = If_w[3 * hf + 1];      wu2 = If_w[3 * hf + 2];
    wr0 = If_w[3 * h];       wr1 = If_w[3 * h + 1];       wr2 = If_w[3 * h + 2];
    we0 = If_w[3 * (S + h)]; we1 = If_w[3 * (S + h) + 1]; we2 = If_w[3 * (S + h) + 2];
  }
  const float vbu = bu[h], vbr = br[h], vbe = be[h];

  const int m0 = tb * 32 + half * 8;        // rows of acc*0
  const int m1 = tb * 32 + 16 + half * 8;   // rows of acc*1
  gru_epilogue(accU0, accR0, accE0, m0, h, wu0, wu1, wu2, wr0, wr1, wr2,
               we0, we1, we2, vbu, vbr, vbe, prev_y, coarse, prev_hidden,
               hidden_f32, hidden_bf);
  gru_epilogue(accU1, accR1, accE1, m1, h, wu0, wu1, wu2, wr0, wr1, wr2,
               we0, we1, we2, vbu, vbr, vbe, prev_y, coarse, prev_hidden,
               hidden_f32, hidden_bf);
}

// ---- Kernel 2: a = relu(hidden[:,koff:koff+S] @ W.T + b) -> bf16 ----------
// One wave per 32(batch) x 16(n) tile, pipelined.  grid = (S/16, B/32).

__global__ void __launch_bounds__(32)
hid_linear_relu(const bf16* __restrict__ Hbf,   // [B,H] bf16
                int koff,                        // 0 (coarse) or S (fine)
                const bf16* __restrict__ W,      // [S,S] bf16
                const float* __restrict__ bias,  // [S]
                bf16* __restrict__ out)          // [B,S] bf16
{
  const int lane = threadIdx.x;
  const int half = lane >> 4;
  const int lid  = lane & 15;
  const int n    = blockIdx.x * 16 + lid;
  const int tb   = blockIdx.y;

  const bf16* arow0 = Hbf + (size_t)(tb * 32 + lid) * H + koff;
  const bf16* arow1 = arow0 + (size_t)16 * H;
  const bf16* wrow  = W + (size_t)n * S;

  floatx8 acc0 = {}, acc1 = {};
  bf16x16 a0 = load_a_frag(arow0, half);
  bf16x16 a1 = load_a_frag(arow1, half);
  bf16x16 w  = load_b_frag(wrow, half);
  for (int kk = 32; kk < S; kk += 32) {
    bf16x16 na0 = load_a_frag(arow0 + kk, half);
    bf16x16 na1 = load_a_frag(arow1 + kk, half);
    bf16x16 nw  = load_b_frag(wrow + kk, half);
    acc0 = wmma_bf16(a0, w, acc0);
    acc1 = wmma_bf16(a1, w, acc1);
    a0 = na0; a1 = na1; w = nw;
  }
  acc0 = wmma_bf16(a0, w, acc0);
  acc1 = wmma_bf16(a1, w, acc1);

  const float bv = bias[n];
  const int m0 = tb * 32 + half * 8;
#pragma unroll
  for (int i = 0; i < 8; ++i) {
    float v0 = acc0[i] + bv; v0 = v0 > 0.0f ? v0 : 0.0f;
    float v1 = acc1[i] + bv; v1 = v1 > 0.0f ? v1 : 0.0f;
    out[(size_t)(m0 + i) * S + n]      = (bf16)v0;
    out[(size_t)(m0 + 16 + i) * S + n] = (bf16)v1;
  }
}

// ---- Kernel 3: logits = a @ W.T + b -> f32 --------------------------------
// One wave per 32(batch) x 16(n) tile, pipelined.  grid = (Q/16, B/32).

__global__ void __launch_bounds__(32)
out_linear(const bf16* __restrict__ Abf,    // [B,S] bf16
           const bf16* __restrict__ W,      // [Q,S] bf16
           const float* __restrict__ bias,  // [Q]
           float* __restrict__ out)         // [B,Q] f32
{
  const int lane = threadIdx.x;
  const int half = lane >> 4;
  const int lid  = lane & 15;
  const int n    = blockIdx.x * 16 + lid;
  const int tb   = blockIdx.y;

  const bf16* arow0 = Abf + (size_t)(tb * 32 + lid) * S;
  const bf16* arow1 = arow0 + (size_t)16 * S;
  const bf16* wrow  = W + (size_t)n * S;

  floatx8 acc0 = {}, acc1 = {};
  bf16x16 a0 = load_a_frag(arow0, half);
  bf16x16 a1 = load_a_frag(arow1, half);
  bf16x16 w  = load_b_frag(wrow, half);
  for (int kk = 32; kk < S; kk += 32) {
    bf16x16 na0 = load_a_frag(arow0 + kk, half);
    bf16x16 na1 = load_a_frag(arow1 + kk, half);
    bf16x16 nw  = load_b_frag(wrow + kk, half);
    acc0 = wmma_bf16(a0, w, acc0);
    acc1 = wmma_bf16(a1, w, acc1);
    a0 = na0; a1 = na1; w = nw;
  }
  acc0 = wmma_bf16(a0, w, acc0);
  acc1 = wmma_bf16(a1, w, acc1);

  const float bv = bias[n];
  const int m0 = tb * 32 + half * 8;
#pragma unroll
  for (int i = 0; i < 8; ++i) {
    out[(size_t)(m0 + i) * Q + n]      = acc0[i] + bv;
    out[(size_t)(m0 + 16 + i) * Q + n] = acc1[i] + bv;
  }
}

// ---------------------------------------------------------------------------

extern "C" void kernel_launch(void* const* d_in, const int* in_sizes, int n_in,
                              void* d_out, int out_size, void* d_ws, size_t ws_size,
                              hipStream_t stream) {
  (void)in_sizes; (void)n_in; (void)out_size; (void)ws_size;

  const float* prev_y      = (const float*)d_in[0];
  const float* prev_hidden = (const float*)d_in[1];
  const float* coarse      = (const float*)d_in[2];
  const float* R_w         = (const float*)d_in[3];
  const float* Ic_w        = (const float*)d_in[4];
  const float* If_w        = (const float*)d_in[5];
  const float* bias_u      = (const float*)d_in[6];
  const float* bias_r      = (const float*)d_in[7];
  const float* bias_e      = (const float*)d_in[8];
  const float* O1_w        = (const float*)d_in[9];
  const float* O1_b        = (const float*)d_in[10];
  const float* O2_w        = (const float*)d_in[11];
  const float* O2_b        = (const float*)d_in[12];
  const float* O3_w        = (const float*)d_in[13];
  const float* O3_b        = (const float*)d_in[14];
  const float* O4_w        = (const float*)d_in[15];
  const float* O4_b        = (const float*)d_in[16];

  // Workspace carve-out (256B aligned): ~94 MB total.
  char* ws = (char*)d_ws;
  size_t off = 0;
  auto carve = [&](size_t elems) -> bf16* {
    bf16* p = (bf16*)(ws + off);
    off += (elems * sizeof(bf16) + 255) & ~(size_t)255;
    return p;
  };
  bf16* ph_bf  = carve((size_t)B * H);   // prev_hidden bf16
  bf16* Rw_bf  = carve((size_t)H3 * H);  // R_w bf16
  bf16* O1_bf  = carve((size_t)S * S);
  bf16* O2_bf  = carve((size_t)Q * S);
  bf16* O3_bf  = carve((size_t)S * S);
  bf16* O4_bf  = carve((size_t)Q * S);
  bf16* hid_bf = carve((size_t)B * H);   // new hidden bf16
  bf16* a1_bf  = carve((size_t)B * S);   // relu(coarse head) bf16
  bf16* a3_bf  = carve((size_t)B * S);   // relu(fine head) bf16

  auto cvt = [&](const float* in, bf16* out, size_t n) {
    int n4 = (int)(n / 4);
    cvt_f32_to_bf16<<<(n4 + 255) / 256, 256, 0, stream>>>(in, out, n4);
  };
  cvt(prev_hidden, ph_bf, (size_t)B * H);
  cvt(R_w,  Rw_bf, (size_t)H3 * H);
  cvt(O1_w, O1_bf, (size_t)S * S);
  cvt(O2_w, O2_bf, (size_t)Q * S);
  cvt(O3_w, O3_bf, (size_t)S * S);
  cvt(O4_w, O4_bf, (size_t)Q * S);

  float* out_coarse = (float*)d_out;                 // [B,Q]
  float* out_fine   = out_coarse + (size_t)B * Q;    // [B,Q]
  float* hidden_out = out_fine + (size_t)B * Q;      // [B,H]

  gru_wmma_kernel<<<dim3(H / 16, B / 32), 32, 0, stream>>>(
      ph_bf, Rw_bf, prev_y, coarse, prev_hidden,
      Ic_w, If_w, bias_u, bias_r, bias_e, hidden_out, hid_bf);

  hid_linear_relu<<<dim3(S / 16, B / 32), 32, 0, stream>>>(
      hid_bf, 0, O1_bf, O1_b, a1_bf);
  hid_linear_relu<<<dim3(S / 16, B / 32), 32, 0, stream>>>(
      hid_bf, S, O3_bf, O3_b, a3_bf);

  out_linear<<<dim3(Q / 16, B / 32), 32, 0, stream>>>(
      a1_bf, O2_bf, O2_b, out_coarse);
  out_linear<<<dim3(Q / 16, B / 32), 32, 0, stream>>>(
      a3_bf, O4_bf, O4_b, out_fine);
}
